// SwinWindowTransformLayer_21792664060087
// MI455X (gfx1250) — matrix operand order repaired and checked by reading, not compiled, not run on hardware
//
#include <hip/hip_runtime.h>
#include <hip/hip_bf16.h>
#include <math.h>

typedef __bf16 bf16;
typedef __attribute__((ext_vector_type(16))) __bf16 v16bf;
typedef __attribute__((ext_vector_type(8)))  __bf16 v8bf;
typedef __attribute__((ext_vector_type(8)))  float  v8f;

#define C_DIM   512
#define C4_DIM  2048
#define ROWS_TOTAL 32768   /* B*N = 8*4096 */
#define M_TILE  64
#define MT      4          /* M_TILE/16: M-tiles per block */
#define KT1     16         /* 512/32  k-steps for GEMM1 */
#define KT2     64         /* 2048/32 k-steps for GEMM2 */

// Branch-free gelu (tanh approx) using hardware exp2/rcp.
__device__ __forceinline__ float gelu_fast(float x) {
    float u = 0.7978845608028654f * x * (1.0f + 0.044715f * x * x);
    u = fminf(fmaxf(u, -15.0f), 15.0f);
    // tanh(u) = (1 - e^{-2u}) / (1 + e^{-2u}),  e^{-2u} = exp2(-2u*log2(e))
    float e = __builtin_amdgcn_exp2f(-2.8853900817779268f * u);
    float t = (1.0f - e) * __builtin_amdgcn_rcpf(1.0f + e);
    return 0.5f * x * (1.0f + t);
}

__device__ __forceinline__ v16bf combine16(v8bf lo, v8bf hi) {
    return __builtin_shufflevector(lo, hi, 0,1,2,3,4,5,6,7,8,9,10,11,12,13,14,15);
}

// A fragment: p points at row_base + kk*32 + (lane>=16 ? 8 : 0); chunks at +0 and +16
__device__ __forceinline__ v16bf load_a_frag(const bf16* p) {
    v8bf lo = *(const v8bf*)(p);
    v8bf hi = *(const v8bf*)(p + 16);
    return combine16(lo, hi);
}

// B fragment: packed layout, per-lane 16 contiguous bf16 (32 bytes)
__device__ __forceinline__ v16bf load_b_frag(const bf16* p) {
    v8bf lo = *(const v8bf*)(p);
    v8bf hi = *(const v8bf*)(p + 8);
    return combine16(lo, hi);
}

// Pack f32 row-major (K,N) weight into bf16 WMMA-B fragment order.
// Flat out index: (((nt*KT + kt)*32 + lane)*16 + j)
//   maps to W[k = kt*32 + (lane>=16?8:0) + (j&7) + (j>=8?16:0)][n = nt*16 + (lane&15)]
__global__ void pack_b_kernel(const float* __restrict__ W, bf16* __restrict__ out,
                              int KT, int N) {
    int tid = blockIdx.x * 256 + threadIdx.x;
    int j   = tid & 15;
    int l   = (tid >> 4) & 31;
    int blk = tid >> 9;
    int kt  = blk % KT;
    int nt  = blk / KT;
    int k = kt * 32 + ((l >> 4) << 3) + (j & 7) + ((j >> 3) << 4);
    int n = nt * 16 + (l & 15);
    out[tid] = (bf16)W[(size_t)k * N + n];
}

__device__ __forceinline__ int shifted_row(int g) {
    int b  = g >> 12;          // /4096
    int nn = g & 4095;
    int hh = nn >> 6, ww = nn & 63;
    return (b << 12) | (((hh + 4) & 63) << 6) | ((ww + 4) & 63);
}

__global__ __launch_bounds__(512, 1)
void swin_mlp_fused(const float* __restrict__ in,
                    const float* __restrict__ gamma,
                    const float* __restrict__ beta,
                    const bf16*  __restrict__ W1p,
                    const float* __restrict__ b1,
                    const bf16*  __restrict__ W2p,
                    const float* __restrict__ b2,
                    float* __restrict__ out) {
    __shared__ bf16 xs[M_TILE * C_DIM];    //  64 KB: LayerNorm'd x (bf16)
    __shared__ bf16 hs[M_TILE * C4_DIM];   // 256 KB: gelu(x@W1+b1) (bf16)

    const int tid  = threadIdx.x;
    const int lane = tid & 31;
    const int wv   = tid >> 5;            // wave id 0..15
    const int row0 = blockIdx.x * M_TILE;
    const int laneN   = lane & 15;
    const int hiHalf  = lane >> 4;        // 0 or 1

    // ---------------- Stage A: shifted add + LayerNorm -> xs ----------------
    {
        const int cbase = lane * 16;
        float4 gm[4], bt[4];
        #pragma unroll
        for (int q = 0; q < 4; ++q) {
            gm[q] = *(const float4*)(gamma + cbase + q * 4);
            bt[q] = *(const float4*)(beta  + cbase + q * 4);
        }
        for (int rr = 0; rr < MT; ++rr) {          // 4 rows per wave
            int m  = wv * MT + rr;
            int g  = row0 + m;
            int g2 = shifted_row(g);
            const float4* p0 = (const float4*)(in + (size_t)g  * C_DIM + cbase);
            const float4* p1 = (const float4*)(in + (size_t)g2 * C_DIM + cbase);
            float x[16];
            float s = 0.f, ss = 0.f;
            #pragma unroll
            for (int q = 0; q < 4; ++q) {
                float4 a = p0[q], b = p1[q];
                x[q*4+0] = a.x + b.x; x[q*4+1] = a.y + b.y;
                x[q*4+2] = a.z + b.z; x[q*4+3] = a.w + b.w;
            }
            #pragma unroll
            for (int i = 0; i < 16; ++i) { s += x[i]; ss += x[i] * x[i]; }
            #pragma unroll
            for (int o = 16; o >= 1; o >>= 1) {
                s  += __shfl_xor(s,  o, 32);
                ss += __shfl_xor(ss, o, 32);
            }
            float mu  = s  * (1.0f / 512.0f);
            float var = ss * (1.0f / 512.0f) - mu * mu;
            float inv = rsqrtf(var + 1e-6f);
            bf16* dst = &xs[m * C_DIM + cbase];
            #pragma unroll
            for (int q = 0; q < 4; ++q) {
                const float gmv[4] = {gm[q].x, gm[q].y, gm[q].z, gm[q].w};
                const float btv[4] = {bt[q].x, bt[q].y, bt[q].z, bt[q].w};
                #pragma unroll
                for (int i = 0; i < 4; ++i)
                    dst[q*4+i] = (bf16)((x[q*4+i] - mu) * inv * gmv[i] + btv[i]);
            }
        }
    }
    __syncthreads();

    // ---------------- Stage B: hs = gelu(xs @ W1 + b1) ----------------
    {
        for (int nt8 = 0; nt8 < 8; ++nt8) {
            int nt = wv * 8 + nt8;               // N-tile 0..127
            v8f acc[MT];
            #pragma unroll
            for (int t = 0; t < MT; ++t) acc[t] = (v8f){};
            const bf16* bp = W1p + (((size_t)nt * KT1) * 32 + lane) * 16;
            #pragma unroll 2
            for (int kk = 0; kk < KT1; ++kk) {
                v16bf bfrag = load_b_frag(bp);
                bp += 32 * 16;
                #pragma unroll
                for (int t = 0; t < MT; ++t) {
                    v16bf a = load_a_frag(&xs[(t * 16 + laneN) * C_DIM + hiHalf * 8 + kk * 32]);
                    acc[t] = __builtin_amdgcn_wmma_f32_16x16x32_bf16(
                        false, a, false, bfrag, (short)0, acc[t], false, false);
                }
            }
            float bb = b1[nt * 16 + laneN];
            #pragma unroll
            for (int t = 0; t < MT; ++t) {
                #pragma unroll
                for (int j = 0; j < 8; ++j) {
                    int mrow = t * 16 + j + hiHalf * 8;
                    hs[mrow * C4_DIM + nt * 16 + laneN] = (bf16)gelu_fast(acc[t][j] + bb);
                }
            }
        }
    }
    __syncthreads();

    // ---------------- Stage C: out = hs @ W2 + b2 + x ----------------
    {
        for (int nt2 = 0; nt2 < 2; ++nt2) {
            int nt = wv * 2 + nt2;               // N-tile 0..31
            v8f acc[MT];
            #pragma unroll
            for (int t = 0; t < MT; ++t) acc[t] = (v8f){};
            const bf16* bp = W2p + (((size_t)nt * KT2) * 32 + lane) * 16;
            #pragma unroll 2
            for (int kk = 0; kk < KT2; ++kk) {
                v16bf bfrag = load_b_frag(bp);
                bp += 32 * 16;
                #pragma unroll
                for (int t = 0; t < MT; ++t) {
                    v16bf a = load_a_frag(&hs[(t * 16 + laneN) * C4_DIM + hiHalf * 8 + kk * 32]);
                    acc[t] = __builtin_amdgcn_wmma_f32_16x16x32_bf16(
                        false, a, false, bfrag, (short)0, acc[t], false, false);
                }
            }
            int n = nt * 16 + laneN;
            float bb = b2[n];
            #pragma unroll
            for (int t = 0; t < MT; ++t) {
                #pragma unroll
                for (int j = 0; j < 8; ++j) {
                    int m  = t * 16 + j + hiHalf * 8;
                    int g  = row0 + m;
                    int g2 = shifted_row(g);
                    float xr = in[(size_t)g * C_DIM + n] + in[(size_t)g2 * C_DIM + n];
                    out[(size_t)g * C_DIM + n] = acc[t][j] + bb + xr;
                }
            }
        }
    }
}

extern "C" void kernel_launch(void* const* d_in, const int* in_sizes, int n_in,
                              void* d_out, int out_size, void* d_ws, size_t ws_size,
                              hipStream_t stream) {
    (void)in_sizes; (void)n_in; (void)out_size; (void)ws_size;
    const float* in_   = (const float*)d_in[0];
    const float* gamma = (const float*)d_in[1];
    const float* beta  = (const float*)d_in[2];
    const float* W1    = (const float*)d_in[3];
    const float* b1    = (const float*)d_in[4];
    const float* W2    = (const float*)d_in[5];
    const float* b2    = (const float*)d_in[6];
    float* out = (float*)d_out;

    bf16* W1p = (bf16*)d_ws;                          // 512*2048 bf16 = 2 MB
    bf16* W2p = W1p + (size_t)C_DIM * C4_DIM;         // 2048*512 bf16 = 2 MB

    const int packElems = C_DIM * C4_DIM;             // 1,048,576
    pack_b_kernel<<<packElems / 256, 256, 0, stream>>>(W1, W1p, KT1, C4_DIM);
    pack_b_kernel<<<packElems / 256, 256, 0, stream>>>(W2, W2p, KT2, C_DIM);

    swin_mlp_fused<<<ROWS_TOTAL / M_TILE, 512, 0, stream>>>(
        in_, gamma, beta, W1p, b1, W2p, b2, out);
}
